// NGCF_29678224016143
// MI455X (gfx1250) — matrix operand order repaired and checked by reading, not compiled, not run on hardware
//
#include <hip/hip_runtime.h>
#include <hip/hip_bf16.h>
#include <stdint.h>

#define N_USERS 100000
#define N_ITEMS 200000
#define N_EDGES 1000000
#define DIM     64
#define NLAYERS 3
#define BATCH   4096
#define SLOPE   0.2f
#define EPSN    1e-12f

typedef __attribute__((ext_vector_type(16))) __bf16 v16bf;
typedef __attribute__((ext_vector_type(8)))  float  v8f;

// ---------------------------------------------------------------------------
// WMMA helpers (V_WMMA_F32_16X16X32_BF16, wave32)
// ---------------------------------------------------------------------------
__device__ __forceinline__ v8f wmma_bf16(v16bf a, v16bf b, v8f c) {
  // 8 args: (neg_a, A, neg_b, B, c_mod, C, reuse_a, reuse_b)
  return __builtin_amdgcn_wmma_f32_16x16x32_bf16(false, a, false, b, (short)0, c,
                                                 false, false);
}

// Load 8 consecutive f32 (32B aligned) into a scalar array.
__device__ __forceinline__ void load8(const float* __restrict__ p, float* o) {
  float4 a = *(const float4*)p;
  float4 b = *(const float4*)(p + 4);
  o[0] = a.x; o[1] = a.y; o[2] = a.z; o[3] = a.w;
  o[4] = b.x; o[5] = b.y; o[6] = b.z; o[7] = b.w;
}

// Build an A fragment (16x32 bf16) from two 8-float chunks.
// Per ISA 7.12.2 (16-bit A 16x32): lane holds row M=lane&15; for lane half h=lane>>4
// VGPR pairs 0..3 hold K = kb*32 + h*8 + {0..7}   -> chunk c0
// VGPR pairs 4..7 hold K = kb*32 + 16 + h*8 + {0..7} -> chunk c1
__device__ __forceinline__ v16bf make_afrag(const float* c0, const float* c1) {
  v16bf a;
#pragma unroll
  for (int t = 0; t < 8; ++t) a[t] = (__bf16)c0[t];
#pragma unroll
  for (int t = 0; t < 8; ++t) a[8 + t] = (__bf16)c1[t];
  return a;
}

// Load a prepacked B fragment (32x16 bf16) for (kb, nt): 8 dwords per lane.
__device__ __forceinline__ v16bf load_bfrag(const uint32_t* __restrict__ Wm,
                                            int kb, int nt, int lane) {
  union { v16bf v; uint32_t u[8]; } r;
  const uint32_t* p = Wm + ((((kb * 4 + nt) * 32) + lane) << 3);
#pragma unroll
  for (int t = 0; t < 8; ++t) r.u[t] = p[t];
  return r.v;
}

__device__ __forceinline__ uint32_t pack_bf16_pair(float lo, float hi) {
  __bf16 l = (__bf16)lo, h = (__bf16)hi;
  uint16_t lu = __builtin_bit_cast(uint16_t, l);
  uint16_t hu = __builtin_bit_cast(uint16_t, h);
  return (uint32_t)lu | ((uint32_t)hu << 16);
}

// ---------------------------------------------------------------------------
// Graph preprocessing
// ---------------------------------------------------------------------------
__global__ void degrees_kernel(const int* __restrict__ src, const int* __restrict__ dst,
                               float* __restrict__ du, float* __restrict__ di, int n) {
  int i = blockIdx.x * 256 + threadIdx.x;
  if (i < n) {
    atomicAdd(&du[src[i]], 1.0f);
    atomicAdd(&di[dst[i]], 1.0f);
  }
}

__global__ void edge_norm_kernel(const int* __restrict__ src, const int* __restrict__ dst,
                                 const float* __restrict__ du, const float* __restrict__ di,
                                 float* __restrict__ nrm, int n) {
  int i = blockIdx.x * 256 + threadIdx.x;
  if (i < n) nrm[i] = rsqrtf(du[src[i]] * di[dst[i]]);
}

// Repack W1/W2 (L x 64 x 64 f32, row-major [k][n]) into per-lane bf16 B fragments.
// Layout per ISA 7.12.2 (16-bit B 32x16): lane: col = lane&15, half h = lane>>4;
// dword j holds K = kb*32 + (j>>2)*16 + h*8 + 2*(j&3) (lo 16b) and K+1 (hi 16b), column nt*16+col.
// Flat index: ((((mat*2+kb)*4+nt)*32+lane)*8+j, mat = l*2 + (0:W1, 1:W2)
__global__ void pack_w_kernel(const float* __restrict__ W1, const float* __restrict__ W2,
                              uint32_t* __restrict__ Wpk) {
  int idx = blockIdx.x * 256 + threadIdx.x;
  if (idx >= 6 * 2048) return;
  int j    = idx & 7;
  int lane = (idx >> 3) & 31;
  int nt   = (idx >> 8) & 3;
  int kb   = (idx >> 10) & 1;
  int mat  = idx >> 11;                 // 0..5
  int l = mat >> 1;
  const float* W = ((mat & 1) ? W2 : W1) + (size_t)l * DIM * DIM;
  int col = lane & 15, h = lane >> 4;
  int n = nt * 16 + col;
  int k = kb * 32 + (j >> 2) * 16 + h * 8 + 2 * (j & 3);
  Wpk[idx] = pack_bf16_pair(W[k * DIM + n], W[(k + 1) * DIM + n]);
}

// ---------------------------------------------------------------------------
// Node transform GEMM: t = h @ W + b   (rows x 64) @ (64 x 64)
// One wave per 16x64 tile; 8 v_wmma_f32_16x16x32_bf16 per tile.
// ---------------------------------------------------------------------------
__global__ void node_gemm_kernel(const float* __restrict__ h, const uint32_t* __restrict__ Wm,
                                 const float* __restrict__ bias, float* __restrict__ t,
                                 int rows) {
  int wave = threadIdx.x >> 5;
  int lane = threadIdx.x & 31;
  int tile = blockIdx.x * 8 + wave;
  int r0 = tile * 16;
  if (r0 >= rows) return;                 // wave-uniform

  int row = lane & 15, hh = lane >> 4;    // A row, lane half
  const float* rp = h + (size_t)(r0 + row) * DIM;
  int c = hh * 8;

  float q0[8], q1[8], q2[8], q3[8];
  load8(rp + c, q0);
  load8(rp + c + 16, q1);
  load8(rp + c + 32, q2);
  load8(rp + c + 48, q3);
  v16bf a0 = make_afrag(q0, q1);          // kb = 0 (K 0..31)
  v16bf a1 = make_afrag(q2, q3);          // kb = 1 (K 32..63)

  int col = lane & 15;
  v8f acc[4];
#pragma unroll
  for (int nt = 0; nt < 4; ++nt) {
    float b = bias[nt * 16 + col];
#pragma unroll
    for (int r = 0; r < 8; ++r) acc[nt][r] = b;
  }

#pragma unroll
  for (int nt = 0; nt < 4; ++nt) {
    v16bf b0 = load_bfrag(Wm, 0, nt, lane);
    v16bf b1 = load_bfrag(Wm, 1, nt, lane);
    acc[nt] = wmma_bf16(a0, b0, acc[nt]);
    acc[nt] = wmma_bf16(a1, b1, acc[nt]);
  }

  // C/D layout: element r -> M = r + 8*hh, N = nt*16 + col
#pragma unroll
  for (int r = 0; r < 8; ++r) {
    size_t orow = (size_t)(r0 + r + 8 * hh) * DIM;
#pragma unroll
    for (int nt = 0; nt < 4; ++nt) t[orow + nt * 16 + col] = acc[nt][r];
  }
}

// ---------------------------------------------------------------------------
// Fused edge-message kernel: one wave per 16 edges.
//   p  = hu[src] * hi[dst]
//   pw = p @ W2 + b2                          (WMMA bf16)
//   atomicAdd hi_acc[dst] += norm*(tu[src]+pw)
//   atomicAdd hu_acc[src] += norm*(ti[dst]+pw)
// ---------------------------------------------------------------------------
__global__ void edge_msg_kernel(const float* __restrict__ hu, const float* __restrict__ hitm,
                                const float* __restrict__ tu, const float* __restrict__ ti,
                                const int* __restrict__ esrc, const int* __restrict__ edst,
                                const float* __restrict__ nrm,
                                const uint32_t* __restrict__ Wm, const float* __restrict__ bias,
                                float* __restrict__ hu_acc, float* __restrict__ hi_acc,
                                int nEdges) {
  __shared__ int   s_src[8][16];
  __shared__ int   s_dst[8][16];
  __shared__ float s_nrm[8][16];

  int wave = threadIdx.x >> 5;
  int lane = threadIdx.x & 31;
  int tile = blockIdx.x * 8 + wave;
  int e0 = tile * 16;
  bool active = (e0 < nEdges);

  if (active && lane < 16) {
    int e = e0 + lane;
    s_src[wave][lane] = esrc[e];
    s_dst[wave][lane] = edst[e];
    s_nrm[wave][lane] = nrm[e];
  }
  __syncthreads();
  if (!active) return;                     // wave-uniform

  int row = lane & 15, hh = lane >> 4;
  int s = s_src[wave][row];
  int d = s_dst[wave][row];
  const float* up = hu + (size_t)s * DIM;
  const float* ip = hitm + (size_t)d * DIM;
  int c = hh * 8;

  float pa[8], pb[8], pc[8], pd[8];
  {
    float ua[8], ub[8], uc[8], ud[8], va[8], vb[8], vc[8], vd[8];
    load8(up + c, ua);       load8(ip + c, va);
    load8(up + c + 16, ub);  load8(ip + c + 16, vb);
    load8(up + c + 32, uc);  load8(ip + c + 32, vc);
    load8(up + c + 48, ud);  load8(ip + c + 48, vd);
#pragma unroll
    for (int t = 0; t < 8; ++t) {
      pa[t] = ua[t] * va[t];
      pb[t] = ub[t] * vb[t];
      pc[t] = uc[t] * vc[t];
      pd[t] = ud[t] * vd[t];
    }
  }
  v16bf a0 = make_afrag(pa, pb);
  v16bf a1 = make_afrag(pc, pd);

  int col = lane & 15;
  v8f acc[4];
#pragma unroll
  for (int nt = 0; nt < 4; ++nt) {
    float b = bias[nt * 16 + col];
#pragma unroll
    for (int r = 0; r < 8; ++r) acc[nt][r] = b;
  }

#pragma unroll
  for (int nt = 0; nt < 4; ++nt) {
    v16bf b0 = load_bfrag(Wm, 0, nt, lane);
    v16bf b1 = load_bfrag(Wm, 1, nt, lane);
    acc[nt] = wmma_bf16(a0, b0, acc[nt]);
    acc[nt] = wmma_bf16(a1, b1, acc[nt]);
  }

  // Epilogue + scatter. Element r of acc corresponds to edge-row r + 8*hh.
#pragma unroll
  for (int r = 0; r < 8; ++r) {
    int er = r + 8 * hh;
    int es = s_src[wave][er];
    int ed = s_dst[wave][er];
    float w = s_nrm[wave][er];
    size_t so = (size_t)es * DIM;
    size_t dof = (size_t)ed * DIM;
#pragma unroll
    for (int nt = 0; nt < 4; ++nt) {
      int n = nt * 16 + col;
      float pw = acc[nt][r];
      float mi = w * (tu[so + n] + pw);
      float mu = w * (ti[dof + n] + pw);
      atomicAdd(&hi_acc[dof + n], mi);
      atomicAdd(&hu_acc[so + n], mu);
    }
  }
}

// ---------------------------------------------------------------------------
// leaky_relu + row L2 normalize. One wave per row (lane handles cols lane, lane+32).
// ---------------------------------------------------------------------------
__global__ void act_norm_kernel(const float* __restrict__ acc, float* __restrict__ out,
                                int rows) {
  int wave = threadIdx.x >> 5;
  int lane = threadIdx.x & 31;
  int row = blockIdx.x * 8 + wave;
  if (row >= rows) return;
  const float* a = acc + (size_t)row * DIM;
  float x0 = a[lane], x1 = a[lane + 32];
  x0 = (x0 > 0.0f) ? x0 : SLOPE * x0;
  x1 = (x1 > 0.0f) ? x1 : SLOPE * x1;
  float s = x0 * x0 + x1 * x1;
#pragma unroll
  for (int m = 16; m >= 1; m >>= 1) s += __shfl_xor(s, m, 32);
  float sc = 1.0f / fmaxf(sqrtf(s), EPSN);
  out[(size_t)row * DIM + lane] = x0 * sc;
  out[(size_t)row * DIM + lane + 32] = x1 * sc;
}

// ---------------------------------------------------------------------------
// Gather batch rows of the current stage into the concatenated outputs.
// d_out = [user(4096x256) | pos(4096x256) | neg(4096x256)], col block = stage*64.
// ---------------------------------------------------------------------------
__global__ void gather_out_kernel(const float* __restrict__ hu, const float* __restrict__ hitm,
                                  const int* __restrict__ users, const int* __restrict__ pos,
                                  const int* __restrict__ neg, float* __restrict__ out,
                                  int stage) {
  int idx = blockIdx.x * 256 + threadIdx.x;
  if (idx >= BATCH * DIM) return;
  int b = idx >> 6, dcol = idx & 63;
  size_t o = (size_t)b * (4 * DIM) + stage * DIM + dcol;
  const size_t SEC = (size_t)BATCH * 4 * DIM;
  out[o]           = hu[(size_t)users[b] * DIM + dcol];
  out[SEC + o]     = hitm[(size_t)pos[b] * DIM + dcol];
  out[2 * SEC + o] = hitm[(size_t)neg[b] * DIM + dcol];
}

// ---------------------------------------------------------------------------
// Host orchestration (graph-capture safe: only async ops on `stream`)
// ---------------------------------------------------------------------------
extern "C" void kernel_launch(void* const* d_in, const int* in_sizes, int n_in,
                              void* d_out, int out_size, void* d_ws, size_t ws_size,
                              hipStream_t stream) {
  const float* user_feat = (const float*)d_in[0];
  const float* item_feat = (const float*)d_in[1];
  const float* W1 = (const float*)d_in[2];
  const float* b1 = (const float*)d_in[3];
  const float* W2 = (const float*)d_in[4];
  const float* b2 = (const float*)d_in[5];
  const int* esrc  = (const int*)d_in[6];
  const int* edst  = (const int*)d_in[7];
  const int* users = (const int*)d_in[8];
  const int* pos   = (const int*)d_in[9];
  const int* neg   = (const int*)d_in[10];
  float* out = (float*)d_out;

  // Workspace carve-up (f32 elements)
  float* ws = (float*)d_ws;
  float* deg_u = ws;                 ws += N_USERS;
  float* deg_i = ws;                 ws += N_ITEMS;
  float* nrm   = ws;                 ws += N_EDGES;
  float* hu    = ws;                 ws += (size_t)N_USERS * DIM;
  float* hitm  = ws;                 ws += (size_t)N_ITEMS * DIM;
  float* tu    = ws;                 ws += (size_t)N_USERS * DIM;
  float* ti    = ws;                 ws += (size_t)N_ITEMS * DIM;
  float* hu_acc = ws;                ws += (size_t)N_USERS * DIM;   // acc buffers adjacent
  float* hi_acc = ws;                ws += (size_t)N_ITEMS * DIM;   // -> single memset
  uint32_t* Wpk = (uint32_t*)ws;     // 6 * 2048 dwords

  // Degrees + per-edge normalization
  hipMemsetAsync(deg_u, 0, (size_t)(N_USERS + N_ITEMS) * sizeof(float), stream);
  degrees_kernel<<<(N_EDGES + 255) / 256, 256, 0, stream>>>(esrc, edst, deg_u, deg_i, N_EDGES);
  edge_norm_kernel<<<(N_EDGES + 255) / 256, 256, 0, stream>>>(esrc, edst, deg_u, deg_i, nrm, N_EDGES);

  // Pack weights into WMMA bf16 B-fragment layout
  pack_w_kernel<<<(6 * 2048 + 255) / 256, 256, 0, stream>>>(W1, W2, Wpk);

  // Stage 0 embeddings = raw features
  hipMemcpyAsync(hu, user_feat, (size_t)N_USERS * DIM * sizeof(float),
                 hipMemcpyDeviceToDevice, stream);
  hipMemcpyAsync(hitm, item_feat, (size_t)N_ITEMS * DIM * sizeof(float),
                 hipMemcpyDeviceToDevice, stream);
  gather_out_kernel<<<(BATCH * DIM + 255) / 256, 256, 0, stream>>>(hu, hitm, users, pos, neg, out, 0);

  const int userTiles = N_USERS / 16;   // 6250 (exact)
  const int itemTiles = N_ITEMS / 16;   // 12500 (exact)
  const int edgeTiles = N_EDGES / 16;   // 62500 (exact)

  for (int l = 0; l < NLAYERS; ++l) {
    const uint32_t* w1m = Wpk + (size_t)(l * 2 + 0) * 2048;
    const uint32_t* w2m = Wpk + (size_t)(l * 2 + 1) * 2048;
    const float* b1l = b1 + l * DIM;
    const float* b2l = b2 + l * DIM;

    node_gemm_kernel<<<(userTiles + 7) / 8, 256, 0, stream>>>(hu, w1m, b1l, tu, N_USERS);
    node_gemm_kernel<<<(itemTiles + 7) / 8, 256, 0, stream>>>(hitm, w1m, b1l, ti, N_ITEMS);

    hipMemsetAsync(hu_acc, 0, (size_t)(N_USERS + N_ITEMS) * DIM * sizeof(float), stream);

    edge_msg_kernel<<<(edgeTiles + 7) / 8, 256, 0, stream>>>(
        hu, hitm, tu, ti, esrc, edst, nrm, w2m, b2l, hu_acc, hi_acc, N_EDGES);

    act_norm_kernel<<<(N_USERS + 7) / 8, 256, 0, stream>>>(hu_acc, hu, N_USERS);
    act_norm_kernel<<<(N_ITEMS + 7) / 8, 256, 0, stream>>>(hi_acc, hitm, N_ITEMS);

    gather_out_kernel<<<(BATCH * DIM + 255) / 256, 256, 0, stream>>>(
        hu, hitm, users, pos, neg, out, l + 1);
  }
}
